// DiscretizedDiagSSMBlock_9680856285213
// MI455X (gfx1250) — matrix operand order repaired and checked by reading, not compiled, not run on hardware
//
#include <hip/hip_runtime.h>
#include <hip/hip_bf16.h>

// ---------------------------------------------------------------------------
// DiscretizedDiagSSM: Y = X @ b_d  (bf16 WMMA GEMM, fp32 accum), then chunked
// linear scan  h_t = a_d * h_{t-1} + y_t  applied in place on d_out.
// B-tile staging uses the gfx1250 Tensor Data Mover (tensor_load_to_lds, the
// 6-arg clang-23 form) with LDS row padding done by the TDM pad engine; falls
// back to load+ds_store if the builtin is unavailable.
// ---------------------------------------------------------------------------

typedef __bf16 bf16_t;
typedef __attribute__((ext_vector_type(16))) __bf16 v16bf;
typedef __attribute__((ext_vector_type(8)))  __bf16 v8bf;
typedef __attribute__((ext_vector_type(8)))  float  v8f;
typedef __attribute__((ext_vector_type(4)))  float  v4f;
typedef unsigned int u32x4 __attribute__((ext_vector_type(4)));
typedef int i32x8 __attribute__((ext_vector_type(8)));
typedef int i32x4 __attribute__((ext_vector_type(4)));

#if defined(__has_builtin)
#if __has_builtin(__builtin_amdgcn_tensor_load_to_lds) && \
    __has_builtin(__builtin_amdgcn_s_wait_tensorcnt)
#define USE_TDM 1
#endif
#endif
#ifndef USE_TDM
#define USE_TDM 0
#endif

static constexpr int kH     = 512;
static constexpr int kB     = 8;
static constexpr int kT     = 8192;
static constexpr int kM     = kB * kT;     // 65536 GEMM rows
static constexpr int kChunk = 128;         // scan chunk length
static constexpr int kC     = kT / kChunk; // 64 chunks
static constexpr int kBH    = kB * kH;     // 4096 channels

// ---- workspace byte offsets ----
static constexpr size_t OFF_AD = 0;                       // H fp32   (a_d)
static constexpr size_t OFF_AL = 2048;                    // H fp32   (a_d^chunk)
static constexpr size_t OFF_BD = 4096;                    // H*H bf16 (b_d)
static constexpr size_t OFF_P  = OFF_BD + (size_t)kH * kH * 2;        // C*B*H fp32
static constexpr size_t OFF_S  = OFF_P  + (size_t)kC * kBH * 4;       // C*B*H fp32

// ---------------------------------------------------------------------------
// Prep: a_d = exp(a*delta), aL = a_d^chunk
// ---------------------------------------------------------------------------
__global__ void ssm_prep_a(const float* __restrict__ a_diag,
                           const float* __restrict__ delta,
                           float* __restrict__ ad, float* __restrict__ aL) {
    int h = blockIdx.x * blockDim.x + threadIdx.x;
    if (h < kH) {
        float da = a_diag[h] * delta[0];
        ad[h] = __expf(da);
        aL[h] = __expf(da * (float)kChunk);
    }
}

// b_d[k][n] = ((exp(da_k)-1)/da_k) * delta * b_mat[k][n], stored bf16
__global__ __launch_bounds__(256) void ssm_prep_b(const float* __restrict__ a_diag,
                                                  const float* __restrict__ b_mat,
                                                  const float* __restrict__ delta,
                                                  bf16_t* __restrict__ bd) {
    int idx = blockIdx.x * 256 + threadIdx.x;   // over H*H
    int k = idx >> 9;                           // H == 512
    float dlt  = delta[0];
    float da   = a_diag[k] * dlt;
    float adk  = __expf(da);
    float coef = ((adk - 1.0f) / da) * dlt;
    bd[idx] = (bf16_t)(coef * b_mat[idx]);
}

// ---------------------------------------------------------------------------
// GEMM: Y[M=65536, N=512] = X[M,K=512] (fp32 -> bf16) @ Bd[K,N] (bf16)
// Block: 256 threads (8 waves). Block tile 128(M) x 64(N); each wave 32x32
// via 2x2 v_wmma_f32_16x16x32_bf16, K stepped by 32 through LDS.
// ---------------------------------------------------------------------------
__global__ __launch_bounds__(256) void ssm_gemm(const float* __restrict__ X,
                                                const bf16_t* __restrict__ Bd,
                                                float* __restrict__ Y) {
    __shared__ bf16_t lA[128][40];  // 32 K + pad 8 (80B row stride, 16B aligned)
    __shared__ bf16_t lB[32][72];   // 64 N + pad 8 (144B row stride)

    const int tid  = threadIdx.x;
    const int lane = tid & 31;
    const int wave = tid >> 5;
    const int wm   = wave & 3;      // wave M quadrant (32 rows)
    const int wn   = wave >> 2;     // wave N half (32 cols)
    const int m0   = blockIdx.y * 128;
    const int n0   = blockIdx.x * 64;

    const int ln15 = lane & 15;
    const int lhi  = lane >> 4;
    const int kb   = lhi * 8;       // A frag K base per lane half (ISA layout)

    // staging assignments
    const int arow = tid >> 1;            // 0..127
    const int acol = (tid & 1) * 16;      // 0 or 16
#if !USE_TDM
    const int brow = tid >> 3;            // 0..31 (K)
    const int bcol = (tid & 7) * 8;       // 0..56 (N)
#endif

    v8f acc[2][2] = {};

    for (int k0 = 0; k0 < kH; k0 += 32) {
        // ---- stage A tile: 128x32 fp32 -> bf16 in LDS (VALU convert path) ----
        const float* gA = X + (size_t)(m0 + arow) * kH + (k0 + acol);
        v4f a0 = *(const v4f*)(gA + 0);
        v4f a1 = *(const v4f*)(gA + 4);
        v4f a2 = *(const v4f*)(gA + 8);
        v4f a3 = *(const v4f*)(gA + 12);
        v8bf c0, c1;
#pragma unroll
        for (int j = 0; j < 4; ++j) {
            c0[j]     = (bf16_t)a0[j];
            c0[4 + j] = (bf16_t)a1[j];
            c1[j]     = (bf16_t)a2[j];
            c1[4 + j] = (bf16_t)a3[j];
        }
        *(v8bf*)&lA[arow][acol]     = c0;
        *(v8bf*)&lA[arow][acol + 8] = c1;

        // ---- stage B tile: 32(K) x 64(N) bf16 ----
#if USE_TDM
        // Tensor Data Mover: one wave-level DMA per block per K-step.
        // D# per ISA ch.8: 2D tensor, data_size=2B, tile 64x32, with LDS pad
        // of 4 DWORDs (8 bf16) after every 32 DWORDs (64 bf16) -> matches
        // the lB[32][72] padded layout consumed below.
        if (wave == 0) {
            unsigned ldsB = (unsigned)(unsigned long long)(void*)&lB[0][0];
            unsigned long long ga =
                (unsigned long long)(const void*)(Bd + (size_t)k0 * kH + n0);
            u32x4 g0;
            g0[0] = 1u;                                     // count=1 (valid)
            g0[1] = ldsB;                                   // lds_addr
            g0[2] = (unsigned)(ga & 0xFFFFFFFFu);           // global_addr lo
            g0[3] = (unsigned)((ga >> 32) & 0x01FFFFFFu)    // global_addr hi
                    | 0x80000000u;                          // type=2 (image)
            i32x8 g1;
            g1[0] = 0x07110000;          // data_size=2B | pad_en | intv=4 | amt=3
            g1[1] = (int)((kH & 0xFFFF) << 16);             // tensor_dim0 lo16
            g1[2] = (int)((kH >> 16) | ((kH & 0xFFFF) << 16)); // d0 hi | d1 lo
            g1[3] = (int)((kH >> 16) | (64 << 16));         // d1 hi | tile_dim0=64
            g1[4] = 32;                                     // tile_dim1=32, dim2=0
            g1[5] = kH;                                     // dim0_stride = 512
            g1[6] = 0;
            g1[7] = 0;
            i32x4 z4 = {0, 0, 0, 0};
            i32x8 z8 = {0, 0, 0, 0, 0, 0, 0, 0};
            __builtin_amdgcn_tensor_load_to_lds(g0, g1, z4, z4, z8, 0);
            __builtin_amdgcn_s_wait_tensorcnt(0);
        }
#else
        *(v8bf*)&lB[brow][bcol] =
            *(const v8bf*)(Bd + (size_t)(k0 + brow) * kH + (n0 + bcol));
#endif

        __syncthreads();

        // ---- build fragments (ISA 7.12.2 layouts) ----
        v16bf fa[2], fb[2];
#pragma unroll
        for (int mi = 0; mi < 2; ++mi) {
            const bf16_t* base = &lA[wm * 32 + mi * 16 + ln15][0];
            v8bf lo = *(const v8bf*)(base + kb);        // K = kb..kb+7
            v8bf hi = *(const v8bf*)(base + kb + 16);   // K = kb+16..kb+23
#pragma unroll
            for (int j = 0; j < 8; ++j) { fa[mi][j] = lo[j]; fa[mi][8 + j] = hi[j]; }
        }
#pragma unroll
        for (int ni = 0; ni < 2; ++ni) {
            const bf16_t* base = &lB[0][wn * 32 + ni * 16 + ln15];  // lane%16 = N, K = kb..kb+7, kb+16..kb+23
#pragma unroll
            for (int j = 0; j < 8; ++j) { fb[ni][j] = base[(kb + j) * 72]; fb[ni][8 + j] = base[(kb + 16 + j) * 72]; }
        }

        // ---- 4 WMMAs ----
#pragma unroll
        for (int mi = 0; mi < 2; ++mi)
#pragma unroll
            for (int ni = 0; ni < 2; ++ni)
                acc[mi][ni] = __builtin_amdgcn_wmma_f32_16x16x32_bf16(
                    false, fa[mi], false, fb[ni], (short)0, acc[mi][ni],
                    false, false);

        __syncthreads();
    }

    // ---- epilogue: C/D layout -> global ----
#pragma unroll
    for (int mi = 0; mi < 2; ++mi)
#pragma unroll
        for (int ni = 0; ni < 2; ++ni)
#pragma unroll
            for (int rr = 0; rr < 8; ++rr) {
                int row = m0 + wm * 32 + mi * 16 + rr + 8 * lhi;
                int col = n0 + wn * 32 + ni * 16 + ln15;
                Y[(size_t)row * kH + col] = acc[mi][ni][rr];
            }
}

// ---------------------------------------------------------------------------
// Scan pass 1: per-(chunk, b, h) local partial with zero carry-in.
// ---------------------------------------------------------------------------
__global__ __launch_bounds__(256) void ssm_scan_partial(const float* __restrict__ Y,
                                                        const float* __restrict__ ad,
                                                        float* __restrict__ P) {
    int gid = blockIdx.x * 256 + threadIdx.x;   // kC * kBH threads
    int c = gid / kBH;
    int r = gid - c * kBH;
    int b = r >> 9;
    int h = r & (kH - 1);
    float a = ad[h];
    float p = 0.0f;
    size_t base = ((size_t)b * kT + (size_t)c * kChunk) * kH + h;
#pragma unroll 8
    for (int i = 0; i < kChunk; ++i)
        p = fmaf(a, p, Y[base + (size_t)i * kH]);
    P[gid] = p;
}

// Pass 2: sequential carry scan across chunks (cheap: 4096 threads x 64).
// S[c] = state entering chunk c; S[0] = h0;  S[c+1] = a^L * S[c] + P[c]
__global__ __launch_bounds__(256) void ssm_scan_carry(const float* __restrict__ h0,
                                                      const float* __restrict__ aL,
                                                      const float* __restrict__ P,
                                                      float* __restrict__ S) {
    int r = blockIdx.x * 256 + threadIdx.x;     // kBH threads
    float s  = h0[r];
    float aa = aL[r & (kH - 1)];
#pragma unroll 4
    for (int c = 0; c < kC; ++c) {
        S[c * kBH + r] = s;
        s = fmaf(aa, s, P[c * kBH + r]);
    }
}

// Pass 3: re-scan each chunk from its true carry-in; write h in place over Y.
__global__ __launch_bounds__(256) void ssm_scan_apply(float* __restrict__ Y,
                                                      const float* __restrict__ ad,
                                                      const float* __restrict__ S) {
    int gid = blockIdx.x * 256 + threadIdx.x;
    int c = gid / kBH;
    int r = gid - c * kBH;
    int b = r >> 9;
    int h = r & (kH - 1);
    float a = ad[h];
    float hs = S[gid];
    size_t base = ((size_t)b * kT + (size_t)c * kChunk) * kH + h;
#pragma unroll 8
    for (int i = 0; i < kChunk; ++i) {
        size_t ix = base + (size_t)i * kH;
        hs = fmaf(a, hs, Y[ix]);
        Y[ix] = hs;
    }
}

// ---------------------------------------------------------------------------
extern "C" void kernel_launch(void* const* d_in, const int* in_sizes, int n_in,
                              void* d_out, int out_size, void* d_ws, size_t ws_size,
                              hipStream_t stream) {
    const float* h0     = (const float*)d_in[0];   // (B,H)
    const float* xs     = (const float*)d_in[1];   // (B,T,H)
    const float* a_diag = (const float*)d_in[2];   // (H,)
    const float* b_mat  = (const float*)d_in[3];   // (H,H)
    const float* delta  = (const float*)d_in[4];   // scalar
    float* out = (float*)d_out;                    // (B,T,H)

    char* ws = (char*)d_ws;
    float*  ad = (float*)(ws + OFF_AD);
    float*  aL = (float*)(ws + OFF_AL);
    bf16_t* bd = (bf16_t*)(ws + OFF_BD);
    float*  P  = (float*)(ws + OFF_P);
    float*  S  = (float*)(ws + OFF_S);

    ssm_prep_a<<<2, 256, 0, stream>>>(a_diag, delta, ad, aL);
    ssm_prep_b<<<(kH * kH) / 256, 256, 0, stream>>>(a_diag, b_mat, delta, bd);

    dim3 gg(kH / 64, kM / 128);                    // (8, 512) blocks
    ssm_gemm<<<gg, 256, 0, stream>>>(xs, bd, out);

    ssm_scan_partial<<<(kC * kBH) / 256, 256, 0, stream>>>(out, ad, P);
    ssm_scan_carry<<<kBH / 256, 256, 0, stream>>>(h0, aL, P, S);
    ssm_scan_apply<<<(kC * kBH) / 256, 256, 0, stream>>>(out, ad, S);
}
